// ODEFunc_90159953478502
// MI455X (gfx1250) — matrix operand order repaired and checked by reading, not compiled, not run on hardware
//
#include <hip/hip_runtime.h>
#include <hip/hip_bf16.h>
#include <math.h>

typedef __bf16 bf16_t;
typedef __attribute__((ext_vector_type(16))) __bf16        v16bf;
typedef __attribute__((ext_vector_type(8)))  float         v8f;
typedef __attribute__((ext_vector_type(4)))  unsigned int  u32x4;

union V16B { v16bf v; u32x4 q[2]; };

constexpr int Dk = 256;    // model dim
constexpr int Hk = 1024;   // hidden dim
constexpr int Ek = 8;      // experts
constexpr int BM = 32;     // batch rows per block (2 M-tiles)
constexpr int XP = Dk + 8; // padded LDS row (bf16 elems): 528B stride, bank-staggered
constexpr int HP = Hk + 8; // padded LDS row: 2064B stride, bank-staggered

// ---------------------------------------------------------------------------
// Branchless tanh. gfx1250 has a hardware V_TANH_F32 transcendental (TRANS32,
// co-executes with XDL WMMA); fall back to exp-identity built from v_exp_f32 +
// v_rcp_f32. No EXEC-mask branches in the GEMM hot loop either way. Result is
// quantized to bf16 immediately, so ~1e-6 relative error is irrelevant.
// ---------------------------------------------------------------------------
__device__ __forceinline__ float fast_tanh(float v) {
#if __has_builtin(__builtin_amdgcn_tanhf)
  return __builtin_amdgcn_tanhf(v);
#else
  float a = fminf(fmaxf(v, -15.f), 15.f);   // avoid inf/inf; tanh(15) == 1 in bf16
  float t = __expf(2.f * a);                // v_exp_f32 (branchless)
  return (t - 1.f) * __builtin_amdgcn_rcpf(t + 1.f);
#endif
}

// ---------------------------------------------------------------------------
// x: f32 -> bf16 (same layout). Also zero-init the expert-active bitmask.
// ---------------------------------------------------------------------------
__global__ void cvt_x_kernel(const float* __restrict__ in, bf16_t* __restrict__ out,
                             int n, unsigned int* __restrict__ mask) {
  int i = blockIdx.x * blockDim.x + threadIdx.x;
  if (i == 0) *mask = 0u;
  if (i < n) out[i] = (bf16_t)in[i];
}

// ---------------------------------------------------------------------------
// Tiled transpose + f32->bf16: in [z][R][C] f32  ->  out [z][C][R] bf16.
// Produces the [N][K] layouts the WMMA B-operand wants as contiguous bytes.
// ---------------------------------------------------------------------------
__global__ void transpose_cvt_kernel(const float* __restrict__ in,
                                     bf16_t* __restrict__ out, int R, int C) {
  __shared__ float tile[32][33];
  size_t base = (size_t)blockIdx.z * R * C;
  int c0 = blockIdx.x * 32, r0 = blockIdx.y * 32;
  int tx = threadIdx.x, ty = threadIdx.y;  // 32 x 8
#pragma unroll
  for (int j = 0; j < 4; ++j) {
    int r = r0 + ty + j * 8;
    tile[ty + j * 8][tx] = in[base + (size_t)r * C + (c0 + tx)];
  }
  __syncthreads();
#pragma unroll
  for (int j = 0; j < 4; ++j) {
    int c = c0 + ty + j * 8;
    out[base + (size_t)c * R + (r0 + tx)] = (bf16_t)tile[tx][ty + j * 8];
  }
}

// ---------------------------------------------------------------------------
// Gating: logits = x @ Wg[0:D] + t*Wg[2D] + bg  (dx_dt block multiplies zeros)
// softmax per row; record which experts exceed threshold anywhere (atomicOr).
// ---------------------------------------------------------------------------
__global__ void gate_kernel(const float* __restrict__ x, const float* __restrict__ Wg,
                            const float* __restrict__ bg, const float* __restrict__ tsc,
                            float* __restrict__ wts, unsigned int* __restrict__ mask,
                            int B) {
  int b = blockIdx.x * blockDim.x + threadIdx.x;
  if (b >= B) return;
  float tv = tsc[0];
  float lg[Ek];
#pragma unroll
  for (int e = 0; e < Ek; ++e) lg[e] = bg[e] + tv * Wg[2 * Dk * Ek + e];
  const float* xr = x + (size_t)b * Dk;
  for (int d = 0; d < Dk; ++d) {
    float xv = xr[d];
    const float* wg = Wg + d * Ek;  // uniform across threads -> scalar loads
#pragma unroll
    for (int e = 0; e < Ek; ++e) lg[e] = fmaf(xv, wg[e], lg[e]);
  }
  float mx = lg[0];
#pragma unroll
  for (int e = 1; e < Ek; ++e) mx = fmaxf(mx, lg[e]);
  float sum = 0.f;
#pragma unroll
  for (int e = 0; e < Ek; ++e) { lg[e] = __expf(lg[e] - mx); sum += lg[e]; }
  float inv = 1.f / sum;
  unsigned int bits = 0u;
#pragma unroll
  for (int e = 0; e < Ek; ++e) {
    float w = lg[e] * inv;
    wts[b * Ek + e] = w;
    if (w > 0.01f) bits |= (1u << e);
  }
  if (bits) atomicOr(mask, bits);
}

// ---------------------------------------------------------------------------
// Fold active mask / uniform fallback into one scale array.
// ---------------------------------------------------------------------------
__global__ void scale_kernel(const float* __restrict__ wts,
                             const unsigned int* __restrict__ mask,
                             float* __restrict__ sc, int BE) {
  int i = blockIdx.x * blockDim.x + threadIdx.x;
  if (i >= BE) return;
  unsigned int m = *mask;
  int e = i & (Ek - 1);
  float s;
  if (m) s = ((m >> e) & 1u) ? wts[i] : 0.f;
  else   s = 1.0f / (float)Ek;
  sc[i] = s;
}

// ---------------------------------------------------------------------------
// Fused MoE, 32 rows (2 M-tiles) per block, 8 waves. Per expert:
//   Phase 1: h = scale*tanh(x @ W1_e + b1_e) -> bf16 LDS tile [32][Hk] (padded)
//            each weight B-operand load feeds 2 WMMAs (2 M-tiles).
//   Phase 2: acc(2Mx2N per wave) += h @ W2_e, accumulated across experts.
// Epilogue: out = acc + sum_e scale[b,e]*b2[e,:]
// Operand addressing per ISA 7.12.2 (16-bit A 16x32, B 32x16, wave32):
//   A  lane l: 16B @ [row=(l&15)][k0+(l>>4)*8] and +16 cols  (row-major src)
//   B  lane l: 16B @ [col=(l&15)][k0+(l>>4)*8] and +16 cols  ([N][K] src)
//   C  vgpr i, lane l: M = i + 8*(l>>4), N = (l&15)
// ---------------------------------------------------------------------------
__global__ __launch_bounds__(256) void moe_fused_kernel(
    const bf16_t* __restrict__ xb,   // [B][D] bf16
    const bf16_t* __restrict__ w1t,  // [E][H][D] bf16 (W1 transposed)
    const bf16_t* __restrict__ w2t,  // [E][D][H] bf16 (W2 transposed)
    const float*  __restrict__ b1,   // [E][H]
    const float*  __restrict__ b2,   // [E][D]
    const float*  __restrict__ sc,   // [B][E] final scales
    float*        __restrict__ out)  // [B][D]
{
  __shared__ bf16_t xLDS[BM * XP];   // ~16.5 KB, padded rows
  __shared__ bf16_t hLDS[BM * HP];   // ~65 KB, padded rows
  __shared__ float  scLDS[BM * Ek];  // 1 KB

  const int tid  = threadIdx.x;
  const int lane = tid & 31;
  const int wave = tid >> 5;   // 0..7
  const int lr   = lane & 15;
  const int hi   = lane >> 4;  // 0/1
  const int m0   = blockIdx.x * BM;

  if (tid < BM * Ek) scLDS[tid] = sc[m0 * Ek + tid];

  // Stage x rows m0..m0+31 into padded LDS (coalesced 16B copies).
  {
    const u32x4* src = (const u32x4*)(xb + (size_t)m0 * Dk);
    for (int i = tid; i < BM * (Dk / 8); i += 256) {
      int row = i / (Dk / 8), c8 = i % (Dk / 8);
      *(u32x4*)(xLDS + row * XP + c8 * 8) = src[i];
    }
  }

  v8f accO[4];  // [mt][nb]: rows mt*16.., cols wave*32 + nb*16; persists over experts
#pragma unroll
  for (int tno = 0; tno < 4; ++tno)
#pragma unroll
    for (int i = 0; i < 8; ++i) accO[tno][i] = 0.f;

  for (int e = 0; e < Ek; ++e) {
    __syncthreads();  // hLDS reuse + staging/scLDS visibility at e==0
    // ---------------- Phase 1 ----------------
    const bf16_t* w1e = w1t + (size_t)e * Hk * Dk;
    const float*  b1e = b1 + e * Hk;
    const bf16_t* a0p = xLDS + (0 * 16 + lr) * XP + hi * 8;
    const bf16_t* a1p = xLDS + (1 * 16 + lr) * XP + hi * 8;
#pragma unroll 1
    for (int nt = 0; nt < 8; ++nt) {        // wave covers h-cols [wave*128,+128)
      const int n0 = wave * 128 + nt * 16;
      v8f h0, h1;
      {
        float bias = b1e[n0 + lr];          // per-column bias, both M-tiles
#pragma unroll
        for (int i = 0; i < 8; ++i) { h0[i] = bias; h1[i] = bias; }
      }
      const bf16_t* wr = w1e + (size_t)(n0 + lr) * Dk + hi * 8;
#pragma unroll
      for (int kc = 0; kc < 8; ++kc) {      // K = 256
        V16B Bv, A0, A1;
        Bv.q[0] = *(const u32x4*)(wr + kc * 32);
        Bv.q[1] = *(const u32x4*)(wr + kc * 32 + 16);
        A0.q[0] = *(const u32x4*)(a0p + kc * 32);
        A0.q[1] = *(const u32x4*)(a0p + kc * 32 + 16);
        A1.q[0] = *(const u32x4*)(a1p + kc * 32);
        A1.q[1] = *(const u32x4*)(a1p + kc * 32 + 16);
        h0 = __builtin_amdgcn_wmma_f32_16x16x32_bf16(false, A0.v, false, Bv.v,
                                                     (short)0, h0, false, false);
        h1 = __builtin_amdgcn_wmma_f32_16x16x32_bf16(false, A1.v, false, Bv.v,
                                                     (short)0, h1, false, false);
      }
#pragma unroll
      for (int i = 0; i < 8; ++i) {
        int ml0 = i + hi * 8;               // rows 0..15
        int ml1 = 16 + ml0;                 // rows 16..31
        hLDS[ml0 * HP + n0 + lr] = (bf16_t)(fast_tanh(h0[i]) * scLDS[ml0 * Ek + e]);
        hLDS[ml1 * HP + n0 + lr] = (bf16_t)(fast_tanh(h1[i]) * scLDS[ml1 * Ek + e]);
      }
    }
    __syncthreads();
    // ---------------- Phase 2 ----------------
    const bf16_t* w2e = w2t + (size_t)e * Dk * Hk;
    const int n0 = wave * 32;
    const bf16_t* wr0 = w2e + (size_t)(n0 + lr) * Hk + hi * 8;
    const bf16_t* wr1 = w2e + (size_t)(n0 + 16 + lr) * Hk + hi * 8;
    const bf16_t* h0p = hLDS + (0 * 16 + lr) * HP + hi * 8;
    const bf16_t* h1p = hLDS + (1 * 16 + lr) * HP + hi * 8;
#pragma unroll 4
    for (int kc = 0; kc < 32; ++kc) {       // K = 1024
      V16B A0, A1, B0, B1;
      A0.q[0] = *(const u32x4*)(h0p + kc * 32);
      A0.q[1] = *(const u32x4*)(h0p + kc * 32 + 16);
      A1.q[0] = *(const u32x4*)(h1p + kc * 32);
      A1.q[1] = *(const u32x4*)(h1p + kc * 32 + 16);
      B0.q[0] = *(const u32x4*)(wr0 + kc * 32);
      B0.q[1] = *(const u32x4*)(wr0 + kc * 32 + 16);
      B1.q[0] = *(const u32x4*)(wr1 + kc * 32);
      B1.q[1] = *(const u32x4*)(wr1 + kc * 32 + 16);
      accO[0] = __builtin_amdgcn_wmma_f32_16x16x32_bf16(false, A0.v, false, B0.v,
                                                        (short)0, accO[0], false, false);
      accO[1] = __builtin_amdgcn_wmma_f32_16x16x32_bf16(false, A0.v, false, B1.v,
                                                        (short)0, accO[1], false, false);
      accO[2] = __builtin_amdgcn_wmma_f32_16x16x32_bf16(false, A1.v, false, B0.v,
                                                        (short)0, accO[2], false, false);
      accO[3] = __builtin_amdgcn_wmma_f32_16x16x32_bf16(false, A1.v, false, B1.v,
                                                        (short)0, accO[3], false, false);
    }
  }

  // ---------------- Epilogue: + sum_e s[b,e]*b2[e,:], store f32 ----------------
  const int n0 = wave * 32;
#pragma unroll
  for (int mt = 0; mt < 2; ++mt) {
#pragma unroll
    for (int i = 0; i < 8; ++i) {
      int ml = mt * 16 + i + hi * 8;
      float e0 = 0.f, e1 = 0.f;
#pragma unroll
      for (int e = 0; e < Ek; ++e) {
        float s = scLDS[ml * Ek + e];
        e0 = fmaf(s, b2[e * Dk + n0 + lr], e0);
        e1 = fmaf(s, b2[e * Dk + n0 + 16 + lr], e1);
      }
      out[(size_t)(m0 + ml) * Dk + n0 + lr]      = accO[mt * 2 + 0][i] + e0;
      out[(size_t)(m0 + ml) * Dk + n0 + 16 + lr] = accO[mt * 2 + 1][i] + e1;
    }
  }
}

// ---------------------------------------------------------------------------
extern "C" void kernel_launch(void* const* d_in, const int* in_sizes, int n_in,
                              void* d_out, int out_size, void* d_ws, size_t ws_size,
                              hipStream_t stream) {
  const float* t  = (const float*)d_in[0];
  const float* x  = (const float*)d_in[1];
  const float* W1 = (const float*)d_in[2];
  const float* b1 = (const float*)d_in[3];
  const float* W2 = (const float*)d_in[4];
  const float* b2 = (const float*)d_in[5];
  const float* Wg = (const float*)d_in[6];
  const float* bg = (const float*)d_in[7];
  float* out = (float*)d_out;

  const int B = in_sizes[1] / Dk;  // 4096

  // Workspace carve-up (all offsets 128B-aligned).
  char* ws = (char*)d_ws;
  const size_t xb_bytes  = (size_t)B * Dk * sizeof(bf16_t);       // 2 MB
  const size_t w1t_bytes = (size_t)Ek * Dk * Hk * sizeof(bf16_t); // 4 MB
  const size_t w2t_bytes = w1t_bytes;                             // 4 MB
  const size_t wts_bytes = (size_t)B * Ek * sizeof(float);        // 128 KB
  bf16_t* xb   = (bf16_t*)(ws);
  bf16_t* w1t  = (bf16_t*)(ws + xb_bytes);
  bf16_t* w2t  = (bf16_t*)(ws + xb_bytes + w1t_bytes);
  float*  wts  = (float*)(ws + xb_bytes + w1t_bytes + w2t_bytes);
  float*  scv  = (float*)(ws + xb_bytes + w1t_bytes + w2t_bytes + wts_bytes);
  unsigned int* mask =
      (unsigned int*)(ws + xb_bytes + w1t_bytes + w2t_bytes + 2 * wts_bytes);

  // 1) Convert x to bf16 (+ zero active mask).
  cvt_x_kernel<<<(B * Dk + 255) / 256, 256, 0, stream>>>(x, xb, B * Dk, mask);

  // 2) Transpose+convert weights into [N][K] bf16 layouts.
  dim3 tb(32, 8);
  transpose_cvt_kernel<<<dim3(Hk / 32, Dk / 32, Ek), tb, 0, stream>>>(W1, w1t, Dk, Hk);
  transpose_cvt_kernel<<<dim3(Dk / 32, Hk / 32, Ek), tb, 0, stream>>>(W2, w2t, Hk, Dk);

  // 3) Gating softmax + active-expert bitmask.
  gate_kernel<<<(B + 255) / 256, 256, 0, stream>>>(x, Wg, bg, t, wts, mask, B);

  // 4) Fold active/fallback rule into scales.
  scale_kernel<<<(B * Ek + 255) / 256, 256, 0, stream>>>(wts, mask, scv, B * Ek);

  // 5) Fused expert GEMMs + mixture (deterministic: register accumulation).
  moe_fused_kernel<<<B / BM, 256, 0, stream>>>(xb, w1t, w2t, b1, b2, scv, out);
}
